// VQVAE_32444182954659
// MI455X (gfx1250) — matrix operand order repaired and checked by reading, not compiled, not run on hardware
//
#include <hip/hip_runtime.h>
#include <math.h>

// ---- model dims ----
#define H_HEADS 8
#define DMODEL  512
#define DLAT_   128
#define NCODES_ 2048
#define NGROUPS_ 64
#define T_ENC_  128
#define BT_     8
#define NSEQ_   16
#define T_DEC_  1024

typedef unsigned short u16;
typedef __attribute__((ext_vector_type(16))) __bf16 bf16x16;
typedef __attribute__((ext_vector_type(8)))  float  f32x8;
typedef unsigned int u32x4 __attribute__((ext_vector_type(4)));
typedef int          i32x4 __attribute__((ext_vector_type(4)));
typedef int          i32x8 __attribute__((ext_vector_type(8)));

union FragBF {
  bf16x16 v;
  unsigned u[8];
  u32x4 q[2];
};

__device__ __forceinline__ u16 f32_to_bf16(float f) {
  union { float f; unsigned u; } x; x.f = f;
  unsigned r = x.u + 0x7FFFu + ((x.u >> 16) & 1u);
  return (u16)(r >> 16);
}
__device__ __forceinline__ f32x8 wmma_bf16(const FragBF& a, const FragBF& b, f32x8 c) {
  return __builtin_amdgcn_wmma_f32_16x16x32_bf16(false, a.v, false, b.v, (short)0, c, false, false);
}

#if defined(__gfx1250__) && __has_builtin(__builtin_amdgcn_tensor_load_to_lds) && \
    __has_builtin(__builtin_amdgcn_s_wait_tensorcnt)
#define USE_TDM 1
#else
#define USE_TDM 0
#endif

#if USE_TDM
// 2D tile DMA: global (bf16, row stride = stride_elems) -> LDS (packed tile_w*tile_h)
// D# group0/group1 per CDNA5 ISA 8.3/8.4; groups 2/3 zero (2D tensor).
// 6-arg builtin form: (g0, g1, g2, g3, g4, cpol)
__device__ __forceinline__ void tdm_load_2d(
    unsigned lds_off, const void* gaddr, unsigned tensor_w, unsigned tensor_h,
    unsigned tile_w, unsigned tile_h, unsigned stride_elems)
{
  unsigned long long ga = (unsigned long long)(size_t)gaddr;
  u32x4 g0;
  g0[0] = 1u;                                            // count=1, user mode
  g0[1] = lds_off;                                       // lds_addr (bytes)
  g0[2] = (unsigned)(ga & 0xffffffffu);                  // global_addr lo
  g0[3] = (unsigned)((ga >> 32) & 0x01ffffffu) | 0x80000000u; // addr hi | type=2
  i32x8 g1;
  g1[0] = (int)(1u << 16);                               // data_size=1 (2 bytes)
  g1[1] = (int)((tensor_w & 0xffffu) << 16);             // tensor_dim0[15:0]
  g1[2] = (int)((tensor_w >> 16) | ((tensor_h & 0xffffu) << 16)); // dim0 hi | dim1 lo
  g1[3] = (int)((tensor_h >> 16) | (tile_w << 16));      // dim1 hi | tile_dim0
  g1[4] = (int)tile_h;                                   // tile_dim1 (tile_dim2=0)
  g1[5] = (int)stride_elems;                             // tensor_dim0_stride lo
  g1[6] = 0;
  g1[7] = 0;
  i32x4 z4 = {0, 0, 0, 0};
  i32x8 z8 = {0, 0, 0, 0, 0, 0, 0, 0};
  __builtin_amdgcn_tensor_load_to_lds(g0, g1, z4, z4, z8, 0);
}
#endif

// =====================================================================
// GEMM: C = A[M,K](bf16) @ Bt[N,K](bf16, pre-transposed) (+bias,+relu)
// outputs: Cf (f32, optional) and/or Cb (bf16, optional)
// block tile 64x128, K-step 32, 8 waves (2x4), TDM double-buffered tiles
// =====================================================================
__global__ __launch_bounds__(256) void gemm_bf16w(
    const u16* __restrict__ A, const u16* __restrict__ Bt,
    const float* __restrict__ bias, float* __restrict__ Cf, u16* __restrict__ Cb,
    int M, int N, int K, int relu)
{
  __shared__ __align__(16) u16 As[2][64][32];
  __shared__ __align__(16) u16 Bs[2][128][32];
  const int tid = threadIdx.x;
  const int lane = tid & 31, wave = tid >> 5;
  const int mlane = lane & 15, hi = lane >> 4;
  const int wm = wave >> 2, wn = wave & 3;
  const int bm = blockIdx.y, bn = blockIdx.x;

  f32x8 acc[2][2] = {};

#if USE_TDM
  if (wave == 0) {
    tdm_load_2d((unsigned)(size_t)&As[0][0][0], A + (size_t)(bm * 64) * K,
                (unsigned)K, (unsigned)M, 32, 64, (unsigned)K);
    tdm_load_2d((unsigned)(size_t)&Bs[0][0][0], Bt + (size_t)(bn * 128) * K,
                (unsigned)K, (unsigned)N, 32, 128, (unsigned)K);
  }
#endif
  int cur = 0;
  for (int k0 = 0; k0 < K; k0 += 32) {
#if USE_TDM
    if (wave == 0) {
      if (k0 + 32 < K) {
        tdm_load_2d((unsigned)(size_t)&As[cur ^ 1][0][0],
                    A + (size_t)(bm * 64) * K + (k0 + 32),
                    (unsigned)K, (unsigned)M, 32, 64, (unsigned)K);
        tdm_load_2d((unsigned)(size_t)&Bs[cur ^ 1][0][0],
                    Bt + (size_t)(bn * 128) * K + (k0 + 32),
                    (unsigned)K, (unsigned)N, 32, 128, (unsigned)K);
        __builtin_amdgcn_s_wait_tensorcnt(2);  // current tile's 2 ops done
      } else {
        __builtin_amdgcn_s_wait_tensorcnt(0);
      }
    }
    __syncthreads();
#else
    for (int i = tid; i < 64 * 32; i += 256)
      As[0][i >> 5][i & 31] = A[(size_t)(bm * 64 + (i >> 5)) * K + k0 + (i & 31)];
    for (int i = tid; i < 128 * 32; i += 256)
      Bs[0][i >> 5][i & 31] = Bt[(size_t)(bn * 128 + (i >> 5)) * K + k0 + (i & 31)];
    __syncthreads();
#endif
    const u16 (*Ac)[32] = As[cur];
    const u16 (*Bc)[32] = Bs[cur];

    FragBF af[2], bb[2];
#pragma unroll
    for (int i = 0; i < 2; i++) {
      int row = wm * 32 + i * 16 + mlane;
      af[i].q[0] = *(const u32x4*)&Ac[row][8 * hi];        // K pairs 0..7 (+8*hi)
      af[i].q[1] = *(const u32x4*)&Ac[row][16 + 8 * hi];   // K pairs 16..23 (+8*hi)
    }
#pragma unroll
    for (int j = 0; j < 2; j++) {
      int col = wn * 32 + j * 16 + mlane;
      bb[j].q[0] = *(const u32x4*)&Bc[col][8 * hi];
      bb[j].q[1] = *(const u32x4*)&Bc[col][16 + 8 * hi];
    }
#pragma unroll
    for (int i = 0; i < 2; i++)
#pragma unroll
      for (int j = 0; j < 2; j++)
        acc[i][j] = wmma_bf16(af[i], bb[j], acc[i][j]);
    __syncthreads();
#if USE_TDM
    cur ^= 1;
#endif
  }

#pragma unroll
  for (int i = 0; i < 2; i++) {
#pragma unroll
    for (int j = 0; j < 2; j++) {
      int col = bn * 128 + wn * 32 + j * 16 + mlane;
      float bv = bias ? bias[col] : 0.0f;
#pragma unroll
      for (int r = 0; r < 8; r++) {
        int row = bm * 64 + wm * 32 + i * 16 + (r + 8 * hi);
        float v = acc[i][j][r] + bv;
        if (relu) v = fmaxf(v, 0.0f);
        if (Cf) Cf[(size_t)row * N + col] = v;
        if (Cb) Cb[(size_t)row * N + col] = f32_to_bf16(v);
      }
    }
  }
}

// =====================================================================
// Flash attention on bf16 qkv (rows t*B+b, cols [Q|K|V], head h at h*64),
// WMMA QK^T + PV, online softmax, bf16 output. 1/sqrt(dh) folded into S.
// =====================================================================
__global__ __launch_bounds__(256) void attn_flash(
    const u16* __restrict__ qkvb, u16* __restrict__ outb,
    int T, int Bb, int causal)
{
  __shared__ __align__(16) u16 Pt[8][16][32];
  const int lane = threadIdx.x & 31, wave = threadIdx.x >> 5;
  const int mlane = lane & 15, hi = lane >> 4;
  const int bh = blockIdx.y, b = bh / H_HEADS, h = bh % H_HEADS;
  const int q0 = blockIdx.x * 128 + wave * 16;

  FragBF qf[2];
  {
    const u16* qb = qkvb + ((size_t)(q0 + mlane) * Bb + b) * 1536 + h * 64;
#pragma unroll
    for (int f = 0; f < 2; f++) {
      qf[f].q[0] = *(const u32x4*)&qb[f * 32 + 8 * hi];
      qf[f].q[1] = *(const u32x4*)&qb[f * 32 + 8 * hi + 16];
    }
  }

  f32x8 oacc[4] = {};
  float mstat[8], lstat[8];
#pragma unroll
  for (int r = 0; r < 8; r++) { mstat[r] = -1e30f; lstat[r] = 0.0f; }

  int kend = causal ? (q0 + 16) : T;
  if (kend > T) kend = T;

  for (int kt = 0; kt < kend; kt += 32) {
    if (kt + 32 < kend)
      __builtin_prefetch(qkvb + ((size_t)(kt + 32 + mlane) * Bb + b) * 1536 + 512 + h * 64, 0, 1);
    // ---- S[16 x 32] = Q . K^T ----
    f32x8 s[2] = {};
#pragma unroll
    for (int sub = 0; sub < 2; sub++) {
      const u16* kb =
          qkvb + ((size_t)(kt + sub * 16 + mlane) * Bb + b) * 1536 + 512 + h * 64;
#pragma unroll
      for (int f = 0; f < 2; f++) {
        FragBF kf;
        kf.q[0] = *(const u32x4*)&kb[f * 32 + 8 * hi];
        kf.q[1] = *(const u32x4*)&kb[f * 32 + 8 * hi + 16];
        s[sub] = wmma_bf16(qf[f], kf, s[sub]);
      }
    }
    // ---- scale + mask + online softmax ----
#pragma unroll
    for (int r = 0; r < 8; r++) {
      int row = q0 + r + 8 * hi;
      float s0 = s[0][r] * 0.125f, s1 = s[1][r] * 0.125f;
      if (causal) {
        if (kt + mlane > row)      s0 = -1e9f;
        if (kt + 16 + mlane > row) s1 = -1e9f;
      }
      float tm = fmaxf(s0, s1);
#pragma unroll
      for (int m = 8; m >= 1; m >>= 1) tm = fmaxf(tm, __shfl_xor(tm, m, 32));
      float nm = fmaxf(mstat[r], tm);
      float corr = __expf(mstat[r] - nm);
      float e0 = __expf(s0 - nm), e1 = __expf(s1 - nm);
      float rs = e0 + e1;
#pragma unroll
      for (int m = 8; m >= 1; m >>= 1) rs += __shfl_xor(rs, m, 32);
      lstat[r] = lstat[r] * corr + rs;
      mstat[r] = nm;
#pragma unroll
      for (int f = 0; f < 4; f++) oacc[f][r] *= corr;
      Pt[wave][r + 8 * hi][mlane]      = f32_to_bf16(e0);
      Pt[wave][r + 8 * hi][16 + mlane] = f32_to_bf16(e1);
    }
    __builtin_amdgcn_wave_barrier();
    FragBF pf;
    pf.q[0] = *(const u32x4*)&Pt[wave][mlane][16 * hi];
    pf.q[1] = *(const u32x4*)&Pt[wave][mlane][16 * hi + 8];
    // ---- O += P . V ----
#pragma unroll
    for (int f = 0; f < 4; f++) {
      FragBF vf;
#pragma unroll
      for (int p = 0; p < 8; p++) {
        int key = kt + 2 * p + 16 * hi;
        const u16* vp =
            qkvb + ((size_t)key * Bb + b) * 1536 + 1024 + h * 64 + f * 16 + mlane;
        unsigned lo = vp[0], hh = vp[(size_t)Bb * 1536];
        vf.u[p] = lo | (hh << 16);
      }
      oacc[f] = wmma_bf16(pf, vf, oacc[f]);
    }
  }

#pragma unroll
  for (int f = 0; f < 4; f++)
#pragma unroll
    for (int r = 0; r < 8; r++) {
      int row = q0 + r + 8 * hi;
      outb[((size_t)row * Bb + b) * DMODEL + h * 64 + f * 16 + mlane] =
          f32_to_bf16(oacc[f][r] / lstat[r]);
    }
}

// =====================================================================
// elementwise / small kernels
// =====================================================================
__device__ __forceinline__ float pe_val(int t, int d) {
  int k = d >> 1;
  float div = __expf((float)(2 * k) * (-9.210340372f / 512.0f));
  float a = (float)t * div;
  return (d & 1) ? __cosf(a) : __sinf(a);
}

__global__ void embed_kernel(const int* __restrict__ toks, const float* __restrict__ emb,
                             float* __restrict__ x, u16* __restrict__ xb,
                             int tshift, size_t n)
{
  size_t idx = (size_t)blockIdx.x * blockDim.x + threadIdx.x;
  if (idx >= n) return;
  int d = (int)(idx & 511);
  size_t row = idx >> 9;
  int t = (int)(row >> tshift);
  int tok = toks[row];
  float v = emb[(size_t)tok * DMODEL + d] * 22.6274170f + pe_val(t, d);
  x[idx] = v;
  if (xb) xb[idx] = f32_to_bf16(v);
}

// residual + layernorm; f32 out (residual stream) + optional bf16 out (GEMM feed)
__global__ __launch_bounds__(256) void add_ln_kernel(
    const float* __restrict__ x, const float* __restrict__ r,
    const float* __restrict__ g, const float* __restrict__ be,
    float* __restrict__ out, u16* __restrict__ outb, int rows)
{
  int row = blockIdx.x * 8 + (threadIdx.x >> 5);
  if (row >= rows) return;
  int lane = threadIdx.x & 31;
  const float* px = x + (size_t)row * DMODEL;
  const float* pr = r + (size_t)row * DMODEL;
  float v[16];
  float s = 0.f;
#pragma unroll
  for (int i = 0; i < 16; i++) { v[i] = px[lane + 32 * i] + pr[lane + 32 * i]; s += v[i]; }
#pragma unroll
  for (int m = 16; m >= 1; m >>= 1) s += __shfl_xor(s, m, 32);
  float mean = s * (1.0f / 512.0f);
  float q = 0.f;
#pragma unroll
  for (int i = 0; i < 16; i++) { float d = v[i] - mean; q += d * d; }
#pragma unroll
  for (int m = 16; m >= 1; m >>= 1) q += __shfl_xor(q, m, 32);
  float rstd = rsqrtf(q * (1.0f / 512.0f) + 1e-5f);
  float* po = out + (size_t)row * DMODEL;
  u16* pb = outb ? outb + (size_t)row * DMODEL : (u16*)0;
#pragma unroll
  for (int i = 0; i < 16; i++) {
    int c = lane + 32 * i;
    float y = (v[i] - mean) * rstd * g[c] + be[c];
    po[c] = y;
    if (pb) pb[c] = f32_to_bf16(y);
  }
}

__global__ void add_kernel(const float* __restrict__ a, const float* __restrict__ b,
                           float* __restrict__ o, u16* __restrict__ ob, size_t n)
{
  size_t i = (size_t)blockIdx.x * blockDim.x + threadIdx.x;
  if (i >= n) return;
  float v = a[i] + b[i];
  o[i] = v;
  ob[i] = f32_to_bf16(v);
}

__global__ void zero_kernel(float* p, int n)
{
  int i = blockIdx.x * blockDim.x + threadIdx.x;
  if (i < n) p[i] = 0.0f;
}

// f32 [K][N] -> bf16 transposed [N][K]
__global__ void cvt_t_kernel(const float* __restrict__ W, u16* __restrict__ Wt,
                             int K, int N)
{
  size_t idx = (size_t)blockIdx.x * blockDim.x + threadIdx.x;
  if (idx >= (size_t)K * N) return;
  int k = (int)(idx % (unsigned)K);
  int n = (int)(idx / (unsigned)K);
  Wt[idx] = f32_to_bf16(W[(size_t)k * N + n]);
}

// grouped VQ, DG=2, one thread per (row, group)
__global__ void vq_kernel(const float* __restrict__ mu, const float* __restrict__ cb,
                          float* __restrict__ z, float* __restrict__ diff)
{
  int idx = blockIdx.x * blockDim.x + threadIdx.x;
  if (idx >= 128 * NGROUPS_) return;
  int row = idx >> 6, grp = idx & 63;
  float g0 = mu[row * DLAT_ + grp * 2 + 0];
  float g1 = mu[row * DLAT_ + grp * 2 + 1];
  float bestd = 3.4e38f; int besti = 0;
  for (int c = 0; c < NCODES_; c++) {
    float d0 = cb[c * 2] - g0, d1 = cb[c * 2 + 1] - g1;
    float dd = d0 * d0 + d1 * d1;
    if (dd < bestd) { bestd = dd; besti = c; }
  }
  float q0 = cb[besti * 2], q1 = cb[besti * 2 + 1];
  z[row * DLAT_ + grp * 2 + 0] = q0;
  z[row * DLAT_ + grp * 2 + 1] = q1;
  float contrib = (q0 - g0) * (q0 - g0) + (q1 - g1) * (q1 - g1);
  atomicAdd(diff, contrib * (2.0f / 16384.0f));
}

// ragged scatter: dseg_b[t,b,:] = bf16(z[b,seg(t),:]), searchsorted(right)-1
__global__ void scatter_kernel(const int* __restrict__ pos, const float* __restrict__ z,
                               u16* __restrict__ dsegb)
{
  size_t idx = (size_t)blockIdx.x * blockDim.x + threadIdx.x;
  if (idx >= (size_t)T_DEC_ * BT_ * DLAT_) return;
  int d = (int)(idx & 127);
  int rb = (int)(idx >> 7);
  int b = rb & 7;
  int t = rb >> 3;
  const int* p = pos + b * (NSEQ_ + 1);
  int cnt = 0;
#pragma unroll
  for (int i = 0; i < NSEQ_ + 1; i++) cnt += (p[i] <= t) ? 1 : 0;
  int seg = cnt - 1;
  float v = 0.0f;
  if (seg >= 0 && seg < NSEQ_) v = z[((size_t)b * NSEQ_ + seg) * DLAT_ + d];
  dsegb[idx] = f32_to_bf16(v);
}

// =====================================================================
extern "C" void kernel_launch(void* const* d_in, const int* in_sizes, int n_in,
                              void* d_out, int out_size, void* d_ws, size_t ws_size,
                              hipStream_t stream)
{
  (void)in_sizes; (void)n_in; (void)out_size; (void)ws_size;

  const int*   enc_inp = (const int*)d_in[0];
  const int*   dec_inp = (const int*)d_in[1];
  const int*   seq_pos = (const int*)d_in[2];
  const float* emb     = (const float*)d_in[3];
  const float* eWqkv = (const float*)d_in[4];
  const float* eWo   = (const float*)d_in[5];
  const float* eln1g = (const float*)d_in[6];
  const float* eln1b = (const float*)d_in[7];
  const float* eln2g = (const float*)d_in[8];
  const float* eln2b = (const float*)d_in[9];
  const float* eW1   = (const float*)d_in[10];
  const float* eb1   = (const float*)d_in[11];
  const float* eW2   = (const float*)d_in[12];
  const float* eb2   = (const float*)d_in[13];
  const float* dWqkv = (const float*)d_in[14];
  const float* dWo   = (const float*)d_in[15];
  const float* dln1g = (const float*)d_in[16];
  const float* dln1b = (const float*)d_in[17];
  const float* dln2g = (const float*)d_in[18];
  const float* dln2b = (const float*)d_in[19];
  const float* dW1   = (const float*)d_in[20];
  const float* db1   = (const float*)d_in[21];
  const float* dW2   = (const float*)d_in[22];
  const float* db2   = (const float*)d_in[23];
  const float* W_mu  = (const float*)d_in[24];
  const float* b_mu  = (const float*)d_in[25];
  const float* cb    = (const float*)d_in[26];
  const float* seg_W = (const float*)d_in[27];
  const float* seg_b = (const float*)d_in[28];
  const float* W_out = (const float*)d_in[29];
  const float* b_out = (const float*)d_in[30];

  float* out       = (float*)d_out;
  float* out_mu    = out;
  float* out_z     = out + 16384;
  float* out_diff  = out + 32768;
  float* out_logit = out + 32769;

  char* cur = (char*)d_ws;
  auto af_ = [&](size_t n) { float* p = (float*)cur; cur += ((n * 4 + 15) & ~(size_t)15); return p; };
  auto ah_ = [&](size_t n) { u16* p = (u16*)cur; cur += ((n * 2 + 15) & ~(size_t)15); return p; };

  float* enc_x = af_((size_t)16384 * 512);
  float* dec_y = af_((size_t)8192 * 512);
  float* dec_x = af_((size_t)8192 * 512);
  float* tmp   = af_((size_t)16384 * 512);
  float* segp  = af_((size_t)8192 * 512);
  u16* enc_xb = ah_((size_t)16384 * 512);
  u16* dec_xb = ah_((size_t)8192 * 512);
  u16* dec_yb = ah_((size_t)8192 * 512);
  u16* qkv_b  = ah_((size_t)16384 * 1536);
  u16* big_b  = ah_((size_t)16384 * 2048);
  u16* attn_b = ah_((size_t)16384 * 512);
  u16* dseg_b = ah_((size_t)8192 * 128);
  u16* eWqkvT = ah_((size_t)4 * 1536 * 512);
  u16* eWoT   = ah_((size_t)4 * 512 * 512);
  u16* eW1T   = ah_((size_t)4 * 2048 * 512);
  u16* eW2T   = ah_((size_t)4 * 512 * 2048);
  u16* dWqkvT = ah_((size_t)6 * 1536 * 512);
  u16* dWoT   = ah_((size_t)6 * 512 * 512);
  u16* dW1T   = ah_((size_t)6 * 2048 * 512);
  u16* dW2T   = ah_((size_t)6 * 512 * 2048);
  u16* WmuT   = ah_((size_t)128 * 512);
  u16* segWT  = ah_((size_t)512 * 128);
  u16* WoutT  = ah_((size_t)1024 * 512);

  auto cvtT = [&](const float* W, u16* Wt, int K, int N) {
    size_t n = (size_t)K * N;
    cvt_t_kernel<<<(unsigned)((n + 255) / 256), 256, 0, stream>>>(W, Wt, K, N);
  };
  auto gemm = [&](const u16* A, const u16* Bt, const float* bias,
                  float* Cf, u16* Cb, int M, int N, int K, int relu) {
    dim3 grid(N / 128, M / 64);
    gemm_bf16w<<<grid, 256, 0, stream>>>(A, Bt, bias, Cf, Cb, M, N, K, relu);
  };

  // ---- weight conversion to transposed bf16 [N][K] ----
  for (int l = 0; l < 4; l++) {
    cvtT(eWqkv + (size_t)l * 512 * 1536, eWqkvT + (size_t)l * 1536 * 512, 512, 1536);
    cvtT(eWo   + (size_t)l * 512 * 512,  eWoT   + (size_t)l * 512 * 512,  512, 512);
    cvtT(eW1   + (size_t)l * 512 * 2048, eW1T   + (size_t)l * 2048 * 512, 512, 2048);
    cvtT(eW2   + (size_t)l * 2048 * 512, eW2T   + (size_t)l * 512 * 2048, 2048, 512);
  }
  for (int l = 0; l < 6; l++) {
    cvtT(dWqkv + (size_t)l * 512 * 1536, dWqkvT + (size_t)l * 1536 * 512, 512, 1536);
    cvtT(dWo   + (size_t)l * 512 * 512,  dWoT   + (size_t)l * 512 * 512,  512, 512);
    cvtT(dW1   + (size_t)l * 512 * 2048, dW1T   + (size_t)l * 2048 * 512, 512, 2048);
    cvtT(dW2   + (size_t)l * 2048 * 512, dW2T   + (size_t)l * 512 * 2048, 2048, 512);
  }
  cvtT(W_mu,  WmuT,  512, 128);
  cvtT(seg_W, segWT, 128, 512);
  cvtT(W_out, WoutT, 512, 1024);

  // ---- embeddings + PE ----
  embed_kernel<<<(16384 * 512) / 256, 256, 0, stream>>>(enc_inp, emb, enc_x, enc_xb, 7,
                                                        (size_t)16384 * 512);
  embed_kernel<<<(8192 * 512) / 256, 256, 0, stream>>>(dec_inp, emb, dec_y, (u16*)0, 3,
                                                       (size_t)8192 * 512);

  // ---- encoder ----
  for (int l = 0; l < 4; l++) {
    gemm(enc_xb, eWqkvT + (size_t)l * 1536 * 512, nullptr, nullptr, qkv_b, 16384, 1536, 512, 0);
    attn_flash<<<dim3(1, 128 * H_HEADS), 256, 0, stream>>>(qkv_b, attn_b, 128, 128, 0);
    gemm(attn_b, eWoT + (size_t)l * 512 * 512, nullptr, tmp, nullptr, 16384, 512, 512, 0);
    add_ln_kernel<<<16384 / 8, 256, 0, stream>>>(enc_x, tmp, eln1g + l * 512,
                                                 eln1b + l * 512, enc_x, enc_xb, 16384);
    gemm(enc_xb, eW1T + (size_t)l * 2048 * 512, eb1 + l * 2048, nullptr, big_b, 16384, 2048, 512, 1);
    gemm(big_b, eW2T + (size_t)l * 512 * 2048, eb2 + l * 512, tmp, nullptr, 16384, 512, 2048, 0);
    add_ln_kernel<<<16384 / 8, 256, 0, stream>>>(enc_x, tmp, eln2g + l * 512,
                                                 eln2b + l * 512, enc_x, enc_xb, 16384);
  }

  // ---- mu (first 128 rows of enc stream are t==0) ----
  gemm(enc_xb, WmuT, b_mu, out_mu, nullptr, 128, 128, 512, 0);

  // ---- VQ + diff ----
  zero_kernel<<<1, 32, 0, stream>>>(out_diff, 1);
  vq_kernel<<<(128 * 64) / 256, 256, 0, stream>>>(out_mu, cb, out_z, out_diff);

  // ---- ragged scatter + segment projection ----
  scatter_kernel<<<(1024 * 8 * 128) / 256, 256, 0, stream>>>(seq_pos, out_z, dseg_b);
  gemm(dseg_b, segWT, seg_b, segp, nullptr, 8192, 512, 128, 0);

  // ---- decoder ----
  for (int l = 0; l < 6; l++) {
    add_kernel<<<(8192 * 512) / 256, 256, 0, stream>>>(dec_y, segp, dec_x, dec_xb,
                                                       (size_t)8192 * 512);
    gemm(dec_xb, dWqkvT + (size_t)l * 1536 * 512, nullptr, nullptr, qkv_b, 8192, 1536, 512, 0);
    attn_flash<<<dim3(1024 / 128, 8 * H_HEADS), 256, 0, stream>>>(qkv_b, attn_b, 1024, 8, 1);
    gemm(attn_b, dWoT + (size_t)l * 512 * 512, nullptr, tmp, nullptr, 8192, 512, 512, 0);
    add_ln_kernel<<<8192 / 8, 256, 0, stream>>>(dec_x, tmp, dln1g + l * 512,
                                                dln1b + l * 512, dec_x, dec_xb, 8192);
    gemm(dec_xb, dW1T + (size_t)l * 2048 * 512, db1 + l * 2048, nullptr, big_b, 8192, 2048, 512, 1);
    gemm(big_b, dW2T + (size_t)l * 512 * 2048, db2 + l * 512, tmp, nullptr, 8192, 512, 2048, 0);
    add_ln_kernel<<<8192 / 8, 256, 0, stream>>>(dec_x, tmp, dln2g + l * 512,
                                                dln2b + l * 512, dec_y, dec_yb, 8192);
  }

  // ---- logits ----
  gemm(dec_yb, WoutT, b_out, out_logit, nullptr, 8192, 1024, 512, 0);
}
